// VGAE_27152783245647
// MI455X (gfx1250) — matrix operand order repaired and checked by reading, not compiled, not run on hardware
//
#include <hip/hip_runtime.h>
#include <hip/hip_bf16.h>

typedef __attribute__((ext_vector_type(16))) _Float16 v16h;
typedef __attribute__((ext_vector_type(8)))  _Float16 v8h;
typedef __attribute__((ext_vector_type(8)))  float    v8f;

#define NNODES 16384
#define LOG2E 1.44269504088896341f

// ---------------- utility kernels ----------------

__global__ __launch_bounds__(256) void k_fill(float* p, float v, int n) {
    int i = blockIdx.x * 256 + threadIdx.x;
    if (i < n) p[i] = v;
}

__global__ __launch_bounds__(256) void k_cvt_f16(const float* s, _Float16* d, int n) {
    int i = blockIdx.x * 256 + threadIdx.x;
    if (i < n) d[i] = (_Float16)s[i];
}

// degree: deg pre-filled with 1.0 (self loop); add 1 per incoming edge
__global__ __launch_bounds__(256) void k_deg(const long long* ei, float* deg, int E) {
    int i = blockIdx.x * 256 + threadIdx.x;
    if (i < E) atomicAdd(&deg[(int)ei[E + i]], 1.0f);
}

__global__ __launch_bounds__(256) void k_dinv(float* deg, int n) {
    int i = blockIdx.x * 256 + threadIdx.x;
    if (i < n) { float d = deg[i]; deg[i] = d > 0.0f ? rsqrtf(d) : 0.0f; }
}

// ---------------- generic WMMA f16 GEMM: C[M,N] = A[M,K] @ B[K,N] ----------------
// one wave -> one 16x16 tile; M,N,K multiples of 16/16/32; grid covers tiles exactly.
__global__ __launch_bounds__(256) void k_gemm_wmma(const _Float16* __restrict__ A,
                                                   const _Float16* __restrict__ B,
                                                   float* __restrict__ C,
                                                   int M, int N, int K) {
    int gid  = blockIdx.x * 256 + threadIdx.x;
    int wave = gid >> 5;
    int lane = threadIdx.x & 31;
    int ntile = N >> 4;
    int ti = wave / ntile, tj = wave % ntile;
    int r  = lane & 15, hi = lane >> 4;
    int arow = ti * 16 + r;     // A row held by this lane
    int bcol = tj * 16 + r;     // B column held by this lane

    v8f acc = {};
    for (int k0 = 0; k0 < K; k0 += 32) {
        // A 16x32 f16 layout: lane(r,hi): elem e = K[k0+hi*16+e]
        const _Float16* ap = A + (size_t)arow * K + k0 + hi * 16;
        v16h a;
#pragma unroll
        for (int e = 0; e < 8; ++e) { a[e] = ap[e]; a[e + 8] = ap[8 + e]; }
        // B 32x16 f16 layout: lane(r,hi): elem e = B[k0 + hi*16 + e][bcol]
        v16h b;
#pragma unroll
        for (int e = 0; e < 16; ++e) b[e] = B[(size_t)(k0 + hi * 16 + e) * N + bcol];
        acc = __builtin_amdgcn_wmma_f32_16x16x32_f16(false, a, false, b, (short)0, acc,
                                                     false, false);
    }
    // C/D: VGPR d -> row hi*8+d, col r
#pragma unroll
    for (int d = 0; d < 8; ++d)
        C[(size_t)(ti * 16 + hi * 8 + d) * N + tj * 16 + r] = acc[d];
}

// ---------------- edge scatter: agg[dst] += dinv[s]*dinv[d] * t[src] ----------------
__global__ __launch_bounds__(256) void k_scatter(const long long* __restrict__ ei,
                                                 const float* __restrict__ dinv,
                                                 const float* __restrict__ t,
                                                 float* __restrict__ agg,
                                                 int E, int fshift) {
    int F = 1 << fshift;
    int total = E << fshift;
    for (int idx = blockIdx.x * 256 + threadIdx.x; idx < total; idx += gridDim.x * 256) {
        int e = idx >> fshift, f = idx & (F - 1);
        int s = (int)ei[e], d = (int)ei[E + e];
        float w = dinv[s] * dinv[d];
        atomicAdd(&agg[((size_t)d << fshift) + f], w * t[((size_t)s << fshift) + f]);
    }
}

// h = relu(agg + dinv^2 * t + b), store f16 for next WMMA stage (F = 64)
__global__ __launch_bounds__(256) void k_bias_relu(const float* __restrict__ agg,
                                                   const float* __restrict__ t,
                                                   const float* __restrict__ dinv,
                                                   const float* __restrict__ b,
                                                   _Float16* __restrict__ h_h, int n) {
    int i = blockIdx.x * 256 + threadIdx.x;
    if (i < n) {
        int node = i >> 6, f = i & 63;
        float di = dinv[node];
        float v = agg[i] + di * di * t[i] + b[f];
        h_h[i] = (_Float16)(v > 0.0f ? v : 0.0f);
    }
}

// finalize mu/logstd (self-loop term + bias, in place in d_out) and build z in f16 (F = 32)
__global__ __launch_bounds__(256) void k_finish(float* __restrict__ mu, float* __restrict__ ls,
                                                const float* __restrict__ tmu,
                                                const float* __restrict__ tls,
                                                const float* __restrict__ dinv,
                                                const float* __restrict__ bmu,
                                                const float* __restrict__ bls,
                                                const float* __restrict__ eps,
                                                _Float16* __restrict__ zh, int n) {
    int i = blockIdx.x * 256 + threadIdx.x;
    if (i < n) {
        int node = i >> 5, f = i & 31;
        float di = dinv[node];
        float w = di * di;
        float m = mu[i] + w * tmu[i] + bmu[f];
        float l = ls[i] + w * tls[i] + bls[f];
        mu[i] = m;
        ls[i] = l;
        float ex = __builtin_amdgcn_exp2f(l * LOG2E);   // exp(logstd)
        zh[i] = (_Float16)(m + eps[i] * ex);
    }
}

// ---------------- decoder: out = sigmoid(z @ z^T), K = 32 -> single WMMA per tile --------
// Each wave computes a 32x32 block (2x2 tiles, 4 independent WMMAs).
__global__ __launch_bounds__(256) void k_decode(const _Float16* __restrict__ zh,
                                                float* __restrict__ out) {
    int gid  = blockIdx.x * 256 + threadIdx.x;
    int wave = gid >> 5;
    int lane = threadIdx.x & 31;
    int ti = wave >> 9;          // 512 super-rows
    int tj = wave & 511;         // 512 super-cols
    int r  = lane & 15, hi = lane >> 4;
    int i0 = ti * 32, j0 = tj * 32;

    // A fragments (rows of z): two 16B chunks per lane per tile row
    const v8h* pa0 = (const v8h*)(zh + (size_t)(i0 + r) * 32 + hi * 16);
    const v8h* pa1 = (const v8h*)(zh + (size_t)(i0 + 16 + r) * 32 + hi * 16);
    v8h a0lo = pa0[0], a0hi = pa0[1];   // +8 halfs = +1 v8h
    v8h a1lo = pa1[0], a1hi = pa1[1];
    v16h a0, a1;
#pragma unroll
    for (int e = 0; e < 8; ++e) {
        a0[e] = a0lo[e]; a0[e + 8] = a0hi[e];
        a1[e] = a1lo[e]; a1[e + 8] = a1hi[e];
    }
    // B fragments: column j of z^T is row j of z -> 16 contiguous halfs at hi*16
    v16h b0 = *(const v16h*)(zh + (size_t)(j0 + r) * 32 + hi * 16);
    v16h b1 = *(const v16h*)(zh + (size_t)(j0 + 16 + r) * 32 + hi * 16);

    v8f c00 = {}, c01 = {}, c10 = {}, c11 = {};
    c00 = __builtin_amdgcn_wmma_f32_16x16x32_f16(false, a0, false, b0, (short)0, c00, false, false);
    c01 = __builtin_amdgcn_wmma_f32_16x16x32_f16(false, a0, false, b1, (short)0, c01, false, false);
    c10 = __builtin_amdgcn_wmma_f32_16x16x32_f16(false, a1, false, b0, (short)0, c10, false, false);
    c11 = __builtin_amdgcn_wmma_f32_16x16x32_f16(false, a1, false, b1, (short)0, c11, false, false);

    const size_t N = (size_t)NNODES;
#pragma unroll
    for (int d = 0; d < 8; ++d) {
        size_t base0 = (size_t)(i0 + hi * 8 + d) * N;
        size_t base1 = (size_t)(i0 + 16 + hi * 8 + d) * N;
        float s;
        s = __builtin_amdgcn_rcpf(1.0f + __builtin_amdgcn_exp2f(-c00[d] * LOG2E));
        __builtin_nontemporal_store(s, out + base0 + j0 + r);
        s = __builtin_amdgcn_rcpf(1.0f + __builtin_amdgcn_exp2f(-c01[d] * LOG2E));
        __builtin_nontemporal_store(s, out + base0 + j0 + 16 + r);
        s = __builtin_amdgcn_rcpf(1.0f + __builtin_amdgcn_exp2f(-c10[d] * LOG2E));
        __builtin_nontemporal_store(s, out + base1 + j0 + r);
        s = __builtin_amdgcn_rcpf(1.0f + __builtin_amdgcn_exp2f(-c11[d] * LOG2E));
        __builtin_nontemporal_store(s, out + base1 + j0 + 16 + r);
    }
}

// ---------------- host launcher ----------------

extern "C" void kernel_launch(void* const* d_in, const int* in_sizes, int n_in,
                              void* d_out, int out_size, void* d_ws, size_t ws_size,
                              hipStream_t stream) {
    const float*     x   = (const float*)d_in[0];
    const long long* ei  = (const long long*)d_in[1];   // int64 per reference
    const float*     W1  = (const float*)d_in[2];
    const float*     b1  = (const float*)d_in[3];
    const float*     Wmu = (const float*)d_in[4];
    const float*     bmu = (const float*)d_in[5];
    const float*     Wls = (const float*)d_in[6];
    const float*     bls = (const float*)d_in[7];
    const float*     eps = (const float*)d_in[8];
    const int N = NNODES;
    const int E = in_sizes[1] / 2;

    // workspace carve-out (~20 MB)
    char* ws = (char*)d_ws;
    size_t off = 0;
    auto carve = [&](size_t bytes) -> char* {
        off = (off + 255) & ~(size_t)255;
        char* p = ws + off;
        off += bytes;
        return p;
    };
    float*    dinv  = (float*)carve((size_t)N * 4);            // deg, then dinv in place
    _Float16* x_h   = (_Float16*)carve((size_t)N * 128 * 2);
    _Float16* W1_h  = (_Float16*)carve(128 * 64 * 2);
    _Float16* Wmu_h = (_Float16*)carve(64 * 32 * 2);
    _Float16* Wls_h = (_Float16*)carve(64 * 32 * 2);
    float*    t1    = (float*)carve((size_t)N * 64 * 4);
    float*    agg1  = (float*)carve((size_t)N * 64 * 4);
    _Float16* h_h   = (_Float16*)carve((size_t)N * 64 * 2);
    float*    tmu   = (float*)carve((size_t)N * 32 * 4);
    float*    tls   = (float*)carve((size_t)N * 32 * 4);
    _Float16* z_h   = (_Float16*)carve((size_t)N * 32 * 2);

    float* out    = (float*)d_out;
    float* mu_out = out + (size_t)N * N;            // [N,32]
    float* ls_out = mu_out + (size_t)N * 32;        // [N,32]

    // init accumulators (harness poisons buffers; must re-zero every call)
    k_fill<<<N / 256, 256, 0, stream>>>(dinv, 1.0f, N);                 // self-loop degree
    k_fill<<<(N * 64) / 256, 256, 0, stream>>>(agg1, 0.0f, N * 64);
    k_fill<<<(N * 64) / 256, 256, 0, stream>>>(mu_out, 0.0f, N * 64);   // mu+logstd contiguous

    // normalization coefficients
    k_deg<<<(E + 255) / 256, 256, 0, stream>>>(ei, dinv, E);
    k_dinv<<<N / 256, 256, 0, stream>>>(dinv, N);

    // f16 copies for the WMMA path
    k_cvt_f16<<<(N * 128) / 256, 256, 0, stream>>>(x, x_h, N * 128);
    k_cvt_f16<<<(128 * 64 + 255) / 256, 256, 0, stream>>>(W1, W1_h, 128 * 64);
    k_cvt_f16<<<(64 * 32 + 255) / 256, 256, 0, stream>>>(Wmu, Wmu_h, 64 * 32);
    k_cvt_f16<<<(64 * 32 + 255) / 256, 256, 0, stream>>>(Wls, Wls_h, 64 * 32);

    // layer 1: t1 = x @ W1 ; scatter ; h = relu(agg + self + b1)
    k_gemm_wmma<<<(N / 16) * (64 / 16) / 8, 256, 0, stream>>>(x_h, W1_h, t1, N, 64, 128);
    k_scatter<<<4096, 256, 0, stream>>>(ei, dinv, t1, agg1, E, 6);
    k_bias_relu<<<(N * 64) / 256, 256, 0, stream>>>(agg1, t1, dinv, b1, h_h, N * 64);

    // layers 2/3: mu & logstd heads (accumulate straight into d_out slices)
    k_gemm_wmma<<<(N / 16) * (32 / 16) / 8, 256, 0, stream>>>(h_h, Wmu_h, tmu, N, 32, 64);
    k_gemm_wmma<<<(N / 16) * (32 / 16) / 8, 256, 0, stream>>>(h_h, Wls_h, tls, N, 32, 64);
    k_scatter<<<4096, 256, 0, stream>>>(ei, dinv, tmu, mu_out, E, 5);
    k_scatter<<<4096, 256, 0, stream>>>(ei, dinv, tls, ls_out, E, 5);

    // reparameterize -> z (f16)
    k_finish<<<(N * 32) / 256, 256, 0, stream>>>(mu_out, ls_out, tmu, tls, dinv,
                                                 bmu, bls, eps, z_h, N * 32);

    // decoder: sigmoid(z z^T), 512x512 waves of 32x32 blocks
    k_decode<<<(512 * 512) / 8, 256, 0, stream>>>(z_h, out);
}
